// RVQ_60498909331608
// MI455X (gfx1250) — matrix operand order repaired and checked by reading, not compiled, not run on hardware
//
#include <hip/hip_runtime.h>
#include <hip/hip_bf16.h>

typedef _Float16 half16 __attribute__((ext_vector_type(16)));
typedef _Float16 half8  __attribute__((ext_vector_type(8)));
typedef float    f32x8  __attribute__((ext_vector_type(8)));

#define B_   8
#define T_   1024
#define D_   256
#define K_   1024
#define L_   4
#define BT_  (B_ * T_)
#define EPSF 1e-12f

// ---------------- small helpers ------------------------------------------

__global__ void k_zero_acc(float* acc) {
    if (threadIdx.x < 4) acc[threadIdx.x] = 0.f;
}

// colnormz[b,d] = || z[b, :, d] ||_2  (normalize over T axis)
__global__ void k_zcolnorm(const float* __restrict__ z, float* __restrict__ colnormz) {
    const int b = blockIdx.x, d = threadIdx.x;
    const float* p = z + (size_t)b * T_ * D_ + d;
    float s = 0.f;
    for (int t = 0; t < T_; ++t) { float v = p[(size_t)t * D_]; s += v * v; }
    colnormz[b * D_ + d] = sqrtf(s);
}

// zn = z / max(colnorm, eps); also emit f16 copy for WMMA A-matrix
__global__ void k_znorm(const float* __restrict__ z, const float* __restrict__ colnormz,
                        float* __restrict__ zn, _Float16* __restrict__ znh) {
    size_t idx = (size_t)blockIdx.x * blockDim.x + threadIdx.x;   // < B*T*D
    int d = (int)(idx & (D_ - 1));
    int b = (int)(idx / ((size_t)T_ * D_));
    float v = z[idx] / fmaxf(colnormz[b * D_ + d], EPSF);
    zn[idx]  = v;
    znh[idx] = (_Float16)v;
}

// cbn[row,:] = cb[row,:] / max(||cb[row,:]||, eps)   row = l*K + k, reduce over D
__global__ void k_cbnorm(const float* __restrict__ cb,
                         float* __restrict__ cbn, _Float16* __restrict__ cbnh) {
    __shared__ float red[D_];
    const int row = blockIdx.x;
    const int d   = threadIdx.x;
    float v = cb[(size_t)row * D_ + d];
    red[d] = v * v;
    __syncthreads();
    for (int s = D_ / 2; s > 0; s >>= 1) {
        if (d < s) red[d] += red[d + s];
        __syncthreads();
    }
    float o = v / fmaxf(sqrtf(red[0]), EPSF);
    cbn[(size_t)row * D_ + d]  = o;
    cbnh[(size_t)row * D_ + d] = (_Float16)o;
}

// ---------------- fused WMMA score GEMM + softmax + label loss -----------
// grid = (BT/16, L), block = 256 (8 waves of 32).
// Wave w computes n-tiles [w*8, w*8+8) of the 16x1024 score tile via
// v_wmma_f32_16x16x32_f16. A-fragments live in registers across all tiles.
// B-fragments use an explicit depth-1 double buffer: fragment ds+1 is loaded
// before the WMMA consuming fragment ds, so the load stays in flight under
// the matrix op instead of draining loadcnt to 0 each step. Scores stage in
// 64 KB LDS, then 16 lanes per row run softmax passes with shfl reductions.
__global__ void __launch_bounds__(256, 1)
k_scores(const _Float16* __restrict__ znh, const _Float16* __restrict__ cbnh,
         const int* __restrict__ iidx, const unsigned char* __restrict__ mask,
         float* __restrict__ accs /* [1] = label sum */) {
    __shared__ float s_lds[16 * 1024];                 // 64 KB score tile

    const int l    = blockIdx.y;
    const int m0   = blockIdx.x * 16;                  // first bt-row of tile
    const int tid  = threadIdx.x;
    const int wave = tid >> 5;
    const int lane = tid & 31;
    const int mrow = lane & 15;                        // A row / B col / C col
    const int hi   = lane >> 4;

    const _Float16* cbl = cbnh + (size_t)l * K_ * D_;

    // Preload all 8 A-fragments (K-dim = D = 256 -> 8 steps of 32) into regs.
    half16 afrag[8];
    #pragma unroll
    for (int ds = 0; ds < 8; ++ds) {
        const _Float16* pa = znh + (size_t)(m0 + mrow) * D_ + ds * 32 + hi * 16;
        half8 alo = *(const half8*)pa;            // K = ds*32 + hi*16 + 0..7
        half8 ahi = *(const half8*)(pa + 8);      // K = ds*32 + hi*16 + 8..15
        half16 a;
        #pragma unroll
        for (int j = 0; j < 8; ++j) { a[j] = alo[j]; a[8 + j] = ahi[j]; }
        afrag[ds] = a;
    }

    // 8 n-tiles per wave -> 64 tiles = K = 1024 columns total.
    for (int nt = wave * 8; nt < wave * 8 + 8; ++nt) {
        const int kidx = nt * 16 + mrow;               // codebook entry (B col)
        const _Float16* pb = cbl + (size_t)kidx * D_ + hi * 16;

        // Depth-1 software pipeline over the 8 d-steps (2 live B buffers).
        half16 b_cur = *(const half16*)pb;
        f32x8 acc = {};
        #pragma unroll
        for (int ds = 0; ds < 8; ++ds) {
            half16 b_nxt;
            if (ds < 7) b_nxt = *(const half16*)(pb + (ds + 1) * 32);
            acc = __builtin_amdgcn_wmma_f32_16x16x32_f16(
                      false, afrag[ds], false, b_cur, (short)0, acc, false, false);
            if (ds < 7) b_cur = b_nxt;
        }

        const int mbase = hi * 8;                      // C rows for this lane
        #pragma unroll
        for (int j = 0; j < 8; ++j)
            s_lds[(mbase + j) * 1024 + nt * 16 + mrow] = acc[j];
    }
    __syncthreads();

    // Softmax / label: 16 lanes per row. p = softmax_k(2*s); label = log(sum exp p) - p_i
    const int r = tid >> 4;            // local row 0..15
    const int g = tid & 15;            // lane-in-row
    const float* srow = s_lds + r * 1024;

    float mx = -3.4e38f;
    for (int k = g; k < 1024; k += 16) mx = fmaxf(mx, 2.f * srow[k]);
    for (int o = 8; o > 0; o >>= 1) mx = fmaxf(mx, __shfl_xor(mx, o, 16));

    float Z = 0.f;
    for (int k = g; k < 1024; k += 16) Z += __expf(2.f * srow[k] - mx);
    for (int o = 8; o > 0; o >>= 1) Z += __shfl_xor(Z, o, 16);
    const float invZ = 1.f / Z;

    float S2 = 0.f;
    for (int k = g; k < 1024; k += 16) S2 += __expf(__expf(2.f * srow[k] - mx) * invZ);
    for (int o = 8; o > 0; o >>= 1) S2 += __shfl_xor(S2, o, 16);

    if (g == 0) {
        const int row = m0 + r;
        if (mask[row]) {
            const int idx = iidx[row * L_ + l];
            const float pi = __expf(2.f * srow[idx] - mx) * invZ;
            atomicAdd(&accs[1], __logf(S2) - pi);
        }
    }
}

// colnormg[l,b,d] = || cbn[l, i[b,:,l], d] ||_2 over T (gathered column norms)
__global__ void k_gcolnorm(const float* __restrict__ cbn, const int* __restrict__ iidx,
                           float* __restrict__ colnormg) {
    const int lb = blockIdx.x;            // l*B + b
    const int l = lb / B_, b = lb % B_;
    const int d = threadIdx.x;
    float s = 0.f;
    for (int t = 0; t < T_; ++t) {
        const int idx = iidx[(b * T_ + t) * L_ + l];
        const float v = cbn[((size_t)(l * K_ + idx)) * D_ + d];
        s += v * v;
        if (t + 1 < T_) {
            const int nidx = iidx[(b * T_ + t + 1) * L_ + l];
            __builtin_prefetch(cbn + ((size_t)(l * K_ + nidx)) * D_ + d, 0, 0);
        }
    }
    colnormg[lb * D_ + d] = sqrtf(s);
}

// z_q (masked sum of per-layer normalized codes) + commit loss + msum
__global__ void k_zq_commit(const float* __restrict__ zn, const float* __restrict__ cbn,
                            const float* __restrict__ colnormg, const int* __restrict__ iidx,
                            const unsigned char* __restrict__ mask,
                            float* __restrict__ out_zq,
                            float* __restrict__ accs /* [0]=commit, [2]=msum */) {
    __shared__ float red[D_];
    const int row = blockIdx.x;           // bt row
    const int d   = threadIdx.x;
    const int b   = row >> 10;            // T = 1024
    const float maskf = mask[row] ? 1.f : 0.f;
    const float znv = zn[(size_t)row * D_ + d];

    float zq = 0.f, dsq = 0.f;
    #pragma unroll
    for (int l = 0; l < L_; ++l) {
        const int idx = iidx[row * L_ + l];
        float c = cbn[((size_t)(l * K_ + idx)) * D_ + d];
        c = c / fmaxf(colnormg[(l * B_ + b) * D_ + d], EPSF);
        zq += c;
        const float diff = znv - c;
        dsq += diff * diff;
    }
    out_zq[(size_t)row * D_ + d] = zq * maskf;

    red[d] = dsq;
    __syncthreads();
    for (int s = D_ / 2; s > 0; s >>= 1) {
        if (d < s) red[d] += red[d + s];
        __syncthreads();
    }
    if (d == 0 && maskf > 0.f) {
        atomicAdd(&accs[0], red[0] * (1.f / D_));   // Σ_l mean_d diff² for this (b,t)
        atomicAdd(&accs[2], 1.f);                   // msum
    }
}

__global__ void k_final(const float* __restrict__ accs, float* __restrict__ out_loss) {
    if (threadIdx.x == 0) {
        const float msum = fmaxf(accs[2], 1.f);
        out_loss[0] = accs[0] / msum;   // commitment_loss
        out_loss[1] = accs[0] / msum;   // codebook_loss (same forward value)
        out_loss[2] = accs[1] / msum;   // label_loss
    }
}

// ---------------- launch ---------------------------------------------------

extern "C" void kernel_launch(void* const* d_in, const int* in_sizes, int n_in,
                              void* d_out, int out_size, void* d_ws, size_t ws_size,
                              hipStream_t stream) {
    const float*         z    = (const float*)d_in[0];          // [B,T,D]
    const int*           iidx = (const int*)d_in[1];            // [B,T,L]
    const unsigned char* mask = (const unsigned char*)d_in[2];  // [B,T] bool
    const float*         cb   = (const float*)d_in[3];          // [L,K,D]

    float* out_zq   = (float*)d_out;                 // BT*D floats
    float* out_loss = out_zq + (size_t)BT_ * D_;     // 3 floats

    // workspace layout (bytes)
    char* ws = (char*)d_ws;
    float*    zn       = (float*)   (ws);                                  //  8 MB
    float*    cbn      = (float*)   (ws + (size_t)8  * 1024 * 1024);       //  4 MB
    _Float16* znh      = (_Float16*)(ws + (size_t)12 * 1024 * 1024);       //  4 MB
    _Float16* cbnh     = (_Float16*)(ws + (size_t)16 * 1024 * 1024);       //  2 MB
    float*    colnormz = (float*)   (ws + (size_t)18 * 1024 * 1024);       //  8 KB
    float*    colnormg = (float*)   (ws + (size_t)18 * 1024 * 1024 + 8192);// 32 KB
    float*    accs     = (float*)   (ws + (size_t)18 * 1024 * 1024 + 8192 + 32768);

    k_zero_acc<<<1, 32, 0, stream>>>(accs);
    k_zcolnorm<<<B_, D_, 0, stream>>>(z, colnormz);
    k_znorm<<<(BT_ * D_) / 256, 256, 0, stream>>>(z, colnormz, zn, znh);
    k_cbnorm<<<L_ * K_, D_, 0, stream>>>(cb, cbn, cbnh);
    k_scores<<<dim3(BT_ / 16, L_), 256, 0, stream>>>(znh, cbnh, iidx, mask, accs);
    k_gcolnorm<<<L_ * B_, D_, 0, stream>>>(cbn, iidx, colnormg);
    k_zq_commit<<<BT_, D_, 0, stream>>>(zn, cbn, colnormg, iidx, mask, out_zq, accs);
    k_final<<<1, 32, 0, stream>>>(accs, out_loss);
}
